// FlowNetC_36730560316062
// MI455X (gfx1250) — matrix operand order repaired and checked by reading, not compiled, not run on hardware
//
#include <hip/hip_runtime.h>
#include <hip/hip_bf16.h>

typedef __attribute__((ext_vector_type(16))) _Float16 v16h;
typedef __attribute__((ext_vector_type(8)))  _Float16 v8h;
typedef __attribute__((ext_vector_type(8)))  float    v8f;

// ---------------------------------------------------------------------------
// f32 -> f16 cast
// ---------------------------------------------------------------------------
__global__ void cast_f32_f16(const float* __restrict__ x, _Float16* __restrict__ y, int n) {
    int i = blockIdx.x * blockDim.x + threadIdx.x;
    if (i < n) y[i] = (_Float16)x[i];
}

// ---------------------------------------------------------------------------
// Pack conv weights (Cout, Cin, 3, 3) f32 -> (Cout, Kpad) f16, K = Cin*9,
// zero padded to Kpad (multiple of 32).
// ---------------------------------------------------------------------------
__global__ void pack_w16(const float* __restrict__ w, _Float16* __restrict__ w16,
                         int Cout, int K, int Kpad) {
    int i = blockIdx.x * blockDim.x + threadIdx.x;
    if (i >= Cout * Kpad) return;
    int m = i / Kpad, k = i - m * Kpad;
    w16[i] = (k < K) ? (_Float16)w[m * K + k] : (_Float16)0.f;
}

// ---------------------------------------------------------------------------
// Implicit-GEMM 3x3 conv, pad=1, stride in {1,2}, f16 in/out, f32 accum via
// v_wmma_f32_16x16x32_f16.
//   block  = 32 output pixels (two 16-pixel N-tiles of one output row)
//            x all Cout (one wave per 16-cout tile; every encoder Cout is a
//            multiple of 16)
//   LDS    = im2col patch, PIXEL-major: col[p][KpadP], KpadP = Kpad + 8 so a
//            lane's B fragment (16 contiguous f16) is two ds_load_b128.
//   A frag = two contiguous 8-f16 runs of the packed weight row (16B loads).
//   Two accumulators per wave share one A fragment (weight reuse).
//   Bias folded into C fragment; leaky-ReLU fused into the f16 store.
// ---------------------------------------------------------------------------
__global__ void conv3x3_wmma_f16(
    const _Float16* __restrict__ in,   // (8, Cin, H, W)
    const _Float16* __restrict__ w16,  // (Cout, Kpad)
    const float*    __restrict__ bias, // (Cout)
    _Float16* __restrict__ out,        // (8, Cout, OH, OW)
    int Cin, int H, int W, int Cout, int OH, int OW, int stride, int Kpad)
{
    extern __shared__ _Float16 col[];  // [32][KpadP]
    const int KpadP = Kpad + 8;

    const int tpr = (OW + 31) >> 5;    // 32-pixel tiles per output row
    int bid    = blockIdx.x;
    int n      = bid / (OH * tpr);
    int rem    = bid - n * (OH * tpr);
    int oh     = rem / tpr;
    int owBase = (rem - oh * tpr) << 5;

    const int K = Cin * 9;

    // cooperative im2col into LDS (zero-pad K tail, OOB pixels, ragged tiles)
    // p fastest across threads -> global reads contiguous across lanes
    for (int idx = threadIdx.x; idx < Kpad * 32; idx += blockDim.x) {
        int p = idx & 31;
        int k = idx >> 5;
        _Float16 v = (_Float16)0.f;
        if (k < K) {
            int ci = k / 9, r = k - ci * 9;
            int ky = r / 3, kx = r - ky * 3;
            int ow = owBase + p;
            int ih = oh * stride - 1 + ky;
            int iw = ow * stride - 1 + kx;
            if (ow < OW && ih >= 0 && ih < H && iw >= 0 && iw < W)
                v = in[(((size_t)n * Cin + ci) * H + ih) * W + iw];
        }
        col[p * KpadP + k] = v;
    }
    __syncthreads();

    const int wave = threadIdx.x >> 5;
    const int lane = threadIdx.x & 31;
    const int l16  = lane & 15;   // A: row m within tile; B/C: column n
    const int hh   = lane >> 4;   // half-wave selector
    const int mt   = wave * 16;   // cout tile base

    // C fragment: element i holds row m = i + 8*hh, col n = l16
    v8f acc0, acc1;
#pragma unroll
    for (int i = 0; i < 8; ++i) { acc0[i] = bias[mt + i + 8 * hh]; acc1[i] = acc0[i]; }

    const _Float16* wrow = w16 + (size_t)(mt + l16) * Kpad;
    const _Float16* b0p  = col + (l16)      * KpadP + 16 * hh;
    const _Float16* b1p  = col + (16 + l16) * KpadP + 16 * hh;

    for (int kk = 0; kk < Kpad; kk += 32) {
        // A (16x32 f16): lane half hh holds K kk+16h .. kk+16h+15, contiguous
        v8h alo = *(const v8h*)(wrow + kk + 16 * hh);
        v8h ahi = *(const v8h*)(wrow + kk + 16 * hh + 8);
        v16h a = __builtin_shufflevector(alo, ahi, 0, 1, 2, 3, 4, 5, 6, 7,
                                                   8, 9, 10, 11, 12, 13, 14, 15);
        // B (32x16 f16): lane half hh holds K kk+16h .. kk+16h+15, contiguous
        v8h b0lo = *(const v8h*)(b0p + kk);
        v8h b0hi = *(const v8h*)(b0p + kk + 8);
        v16h b0 = __builtin_shufflevector(b0lo, b0hi, 0, 1, 2, 3, 4, 5, 6, 7,
                                                      8, 9, 10, 11, 12, 13, 14, 15);
        v8h b1lo = *(const v8h*)(b1p + kk);
        v8h b1hi = *(const v8h*)(b1p + kk + 8);
        v16h b1 = __builtin_shufflevector(b1lo, b1hi, 0, 1, 2, 3, 4, 5, 6, 7,
                                                      8, 9, 10, 11, 12, 13, 14, 15);
        acc0 = __builtin_amdgcn_wmma_f32_16x16x32_f16(
            false, a, false, b0, (short)0, acc0, false, false);
        acc1 = __builtin_amdgcn_wmma_f32_16x16x32_f16(
            false, a, false, b1, (short)0, acc1, false, false);
    }

#pragma unroll
    for (int g = 0; g < 2; ++g) {
        int ow = owBase + g * 16 + l16;
        if (ow < OW) {
            const v8f& acc = g ? acc1 : acc0;
#pragma unroll
            for (int i = 0; i < 8; ++i) {
                int co = mt + i + 8 * hh;
                float v = acc[i];
                v = (v >= 0.f) ? v : 0.1f * v;             // fused leaky-ReLU
                out[(((size_t)n * Cout + co) * OH + oh) * OW + ow] = (_Float16)v;
            }
        }
    }
}

// ---------------------------------------------------------------------------
// Correlation (81 shifts, dx,dy in [-2,6]) + leaky-ReLU.  f1,f2 are f16
// (8,96,4,4); out f32 (8,81,4,4).
// ---------------------------------------------------------------------------
__global__ void corr_lrelu(const _Float16* __restrict__ f1, const _Float16* __restrict__ f2,
                           float* __restrict__ out, int N, int C, int H, int W) {
    int i = blockIdx.x * blockDim.x + threadIdx.x;
    int total = N * 81 * H * W;
    if (i >= total) return;
    int x = i % W; int t = i / W;
    int y = t % H; t /= H;
    int s = t % 81; int n = t / 81;
    int dx = s / 9 - 2;
    int dy = s % 9 - 2;
    int ys = y - dy, xs = x - dx;
    float acc = 0.f;
    if (ys >= 0 && ys < H && xs >= 0 && xs < W) {
        for (int c = 0; c < C; ++c)
            acc += (float)f1[((n * C + c) * H + y) * W + x] *
                   (float)f2[((n * C + c) * H + ys) * W + xs];
    }
    acc /= (float)C;
    out[i] = (acc >= 0.f) ? acc : 0.1f * acc;
}

// ---------------------------------------------------------------------------
// 1x1 conv (f16 in, f32 out) + bias + leaky-ReLU  (conv_redir)
// ---------------------------------------------------------------------------
__global__ void conv1x1_lrelu(const _Float16* __restrict__ in, const float* __restrict__ w,
                              const float* __restrict__ b, float* __restrict__ out,
                              int N, int Cin, int Cout, int HW) {
    int i = blockIdx.x * blockDim.x + threadIdx.x;
    if (i >= N * Cout * HW) return;
    int p = i % HW; int t = i / HW;
    int co = t % Cout; int n = t / Cout;
    float acc = b[co];
    for (int ci = 0; ci < Cin; ++ci)
        acc += w[co * Cin + ci] * (float)in[((size_t)(n * Cin + ci)) * HW + p];
    out[i] = (acc >= 0.f) ? acc : 0.1f * acc;
}

// ---------------------------------------------------------------------------
// Direct 3x3 conv, pad=1, stride=1, f32, two virtually-concatenated inputs
// (A: CA channels, then B: CB channels), bias + leaky-ReLU.
// ---------------------------------------------------------------------------
__global__ void conv3x3_direct2(const float* __restrict__ A, int CA,
                                const float* __restrict__ B, int CB,
                                const float* __restrict__ w, const float* __restrict__ bias,
                                float* __restrict__ out, int N, int H, int W, int Cout) {
    int i = blockIdx.x * blockDim.x + threadIdx.x;
    if (i >= N * Cout * H * W) return;
    int x = i % W; int t = i / W;
    int y = t % H; t /= H;
    int co = t % Cout; int n = t / Cout;
    int Cin = CA + CB;
    float acc = bias[co];
    for (int ci = 0; ci < Cin; ++ci) {
        const float* S; int c, Cs;
        if (ci < CA) { S = A; c = ci;      Cs = CA; }
        else         { S = B; c = ci - CA; Cs = CB; }
        const float* wp = w + ((size_t)co * Cin + ci) * 9;
        const float* sp = S + ((size_t)n * Cs + c) * H * W;
        for (int ky = 0; ky < 3; ++ky) {
            int iy = y - 1 + ky;
            if (iy < 0 || iy >= H) continue;
            for (int kx = 0; kx < 3; ++kx) {
                int ix = x - 1 + kx;
                if (ix < 0 || ix >= W) continue;
                acc += wp[ky * 3 + kx] * sp[iy * W + ix];
            }
        }
    }
    out[i] = (acc >= 0.f) ? acc : 0.1f * acc;
}

// ---------------------------------------------------------------------------
// 4x4 stride-2 transposed conv (deconv) for flow (2 -> 2 channels), no bias.
// Matches reference: k = transpose(w,(1,0,2,3))[:,:,::-1,::-1], lhs_dilation=2,
// pad ((2,2),(2,2)).  out (N,2,2H,2W).
// ---------------------------------------------------------------------------
__global__ void deconv4x4_flow(const float* __restrict__ flow, const float* __restrict__ w,
                               float* __restrict__ out, int N, int H, int W) {
    int OH = 2 * H, OW = 2 * W;
    int i = blockIdx.x * blockDim.x + threadIdx.x;
    if (i >= N * 2 * OH * OW) return;
    int x = i % OW; int t = i / OW;
    int y = t % OH; t /= OH;
    int o = t % 2; int n = t / 2;
    float acc = 0.f;
    for (int ic = 0; ic < 2; ++ic) {
        for (int ky = 0; ky < 4; ++ky) {
            int yy = y - 2 + ky;
            if (yy < 0 || (yy & 1)) continue;
            int sy = yy >> 1;
            if (sy >= H) continue;
            for (int kx = 0; kx < 4; ++kx) {
                int xx = x - 2 + kx;
                if (xx < 0 || (xx & 1)) continue;
                int sx = xx >> 1;
                if (sx >= W) continue;
                // w layout (i, o, 4, 4); kernel flipped both ways
                acc += w[((ic * 2 + o) * 4 + (3 - ky)) * 4 + (3 - kx)] *
                       flow[((n * 2 + ic) * H + sy) * W + sx];
            }
        }
    }
    out[i] = acc;
}

// ---------------------------------------------------------------------------
// Align-corners bilinear resize, with output scale factor.
// ---------------------------------------------------------------------------
__global__ void bilinear_ac_k(const float* __restrict__ in, float* __restrict__ out,
                              int N, int C, int H, int W, int OH, int OW, float scale) {
    int i = blockIdx.x * blockDim.x + threadIdx.x;
    if (i >= N * C * OH * OW) return;
    int ox = i % OW; int t = i / OW;
    int oy = t % OH; t /= OH;
    int c = t % C; int n = t / C;
    float fy = (OH > 1) ? (float)oy * (float)(H - 1) / (float)(OH - 1) : 0.f;
    float fx = (OW > 1) ? (float)ox * (float)(W - 1) / (float)(OW - 1) : 0.f;
    int y0 = (int)fy, x0 = (int)fx;
    int y1 = y0 + 1 < H ? y0 + 1 : H - 1;
    int x1 = x0 + 1 < W ? x0 + 1 : W - 1;
    float wy = fy - (float)y0, wx = fx - (float)x0;
    const float* p = in + ((size_t)n * C + c) * H * W;
    float top = p[y0 * W + x0] * (1.f - wx) + p[y0 * W + x1] * wx;
    float bot = p[y1 * W + x0] * (1.f - wx) + p[y1 * W + x1] * wx;
    out[i] = (top * (1.f - wy) + bot * wy) * scale;
}

// ---------------------------------------------------------------------------
// Host driver
// ---------------------------------------------------------------------------
extern "C" void kernel_launch(void* const* d_in, const int* in_sizes, int n_in,
                              void* d_out, int out_size, void* d_ws, size_t ws_size,
                              hipStream_t stream) {
    (void)in_sizes; (void)n_in; (void)out_size; (void)ws_size;

    char* ws = (char*)d_ws;
    auto F16 = [&](size_t off) { return (_Float16*)(ws + off); };
    auto F32 = [&](size_t off) { return (float*)(ws + off); };

    // ---- workspace layout (bytes) ----
    // encoder region
    const size_t XH_OFF  = 0;            // 8*3*1024*1024 f16  = 50,331,648 B
    const size_t A_OFF   = 50331648;     // ping activation (<= 33.5M f16 = 67,108,864 B)
    const size_t B_OFF   = 117440512;    // pong activation
    const size_t F1_OFF  = 184549376;    // 8*96*4*4 f16
    const size_t F2_OFF  = 184573952;
    const size_t W16_OFF = 184598528;    // packed f16 weights (~685 KB)
    // decoder region (reuses [0, 184 MB) after encoder is done)
    const size_t FEAT0_OFF  = 0;          // 8*256*64*64 f32 = 33,554,432 B
    const size_t FEAT1_OFF  = 33554432;
    const size_t UPFLOW_OFF = 67108864;   // 8*2*64*64 f32
    const size_t FLOW0_OFF  = 68157440;
    const size_t FLOW1_OFF  = 69206016;
    const size_t CORR_OFF   = 70254592;   // 8*81*16 f32
    const size_t REDIR_OFF  = 71303168;   // 8*32*16 f32

    const float* x[2] = {(const float*)d_in[0], (const float*)d_in[1]};

    struct L { int wi, bi, ci, co, s; };
    static const L enc[12] = {
        { 2,  3,  3, 16, 2}, { 4,  5, 16, 16, 1}, { 6,  7, 16, 16, 2},
        { 8,  9, 16, 32, 2}, {10, 11, 32, 32, 1}, {12, 13, 32, 32, 2},
        {14, 15, 32, 64, 2}, {16, 17, 64, 64, 1}, {18, 19, 64, 64, 2},
        {20, 21, 64, 96, 2}, {22, 23, 96, 96, 1}, {24, 25, 96, 96, 2}};

    // ---- pack encoder weights to f16 (Cout x Kpad) ----
    size_t woff[12]; size_t wacc = 0;
    int kpad[12];
    for (int l = 0; l < 12; ++l) {
        kpad[l] = ((enc[l].ci * 9 + 31) / 32) * 32;
        woff[l] = wacc;
        wacc += (size_t)enc[l].co * kpad[l];
    }
    for (int l = 0; l < 12; ++l) {
        int tot = enc[l].co * kpad[l];
        pack_w16<<<(tot + 255) / 256, 256, 0, stream>>>(
            (const float*)d_in[enc[l].wi], F16(W16_OFF) + woff[l],
            enc[l].co, enc[l].ci * 9, kpad[l]);
    }

    // ---- encoder (both images) ----
    for (int e = 0; e < 2; ++e) {
        int nIn = 8 * 3 * 1024 * 1024;
        cast_f32_f16<<<(nIn + 255) / 256, 256, 0, stream>>>(x[e], F16(XH_OFF), nIn);
        const _Float16* cur = F16(XH_OFF);
        int H = 1024, W = 1024;
        for (int l = 0; l < 12; ++l) {
            const L& P = enc[l];
            int OH = (P.s == 2) ? H / 2 : H;
            int OW = (P.s == 2) ? W / 2 : W;
            _Float16* outp = (l == 11) ? F16(e ? F2_OFF : F1_OFF)
                                       : F16((l & 1) ? B_OFF : A_OFF);
            int tpr = (OW + 31) / 32;
            dim3 grid(8u * (unsigned)OH * (unsigned)tpr);
            dim3 blk(32u * (unsigned)(P.co / 16));
            size_t shmem = (size_t)(kpad[l] + 8) * 32 * sizeof(_Float16);
            conv3x3_wmma_f16<<<grid, blk, shmem, stream>>>(
                cur, F16(W16_OFF) + woff[l], (const float*)d_in[P.bi], outp,
                P.ci, H, W, P.co, OH, OW, P.s, kpad[l]);
            cur = outp; H = OH; W = OW;
        }
    }

    // ---- correlation + redir + conv3_1 + predict_flow6 (4x4 spatial) ----
    const _Float16* f1 = F16(F1_OFF);
    const _Float16* f2 = F16(F2_OFF);
    float* corr  = F32(CORR_OFF);
    float* redir = F32(REDIR_OFF);
    corr_lrelu<<<(8 * 81 * 16 + 255) / 256, 256, 0, stream>>>(f1, f2, corr, 8, 96, 4, 4);
    conv1x1_lrelu<<<(8 * 32 * 16 + 255) / 256, 256, 0, stream>>>(
        f1, (const float*)d_in[26], (const float*)d_in[27], redir, 8, 96, 32, 16);

    float* feat = F32(FEAT0_OFF);
    conv3x3_direct2<<<(8 * 256 * 16 + 255) / 256, 256, 0, stream>>>(
        redir, 32, corr, 81, (const float*)d_in[28], (const float*)d_in[29],
        feat, 8, 4, 4, 256);

    float* flow = F32(FLOW0_OFF);
    conv3x3_direct2<<<(8 * 2 * 16 + 255) / 256, 256, 0, stream>>>(
        feat, 256, nullptr, 0, (const float*)d_in[30], (const float*)d_in[31],
        flow, 8, 4, 4, 2);

    // ---- flow pyramid: levels 5,4,3,2 ----
    float* featAlt = F32(FEAT1_OFF);
    float* flowAlt = F32(FLOW1_OFF);
    float* upflow  = F32(UPFLOW_OFF);
    int H = 4;
    for (int li = 0; li < 4; ++li) {
        int base = 32 + li * 3;  // up{l+1}to{l}_w, predict_flow{l}_w, predict_flow{l}_b
        int OH = 2 * H;
        deconv4x4_flow<<<(8 * 2 * OH * OH + 255) / 256, 256, 0, stream>>>(
            flow, (const float*)d_in[base], upflow, 8, H, H);
        bilinear_ac_k<<<(8 * 256 * OH * OH + 255) / 256, 256, 0, stream>>>(
            feat, featAlt, 8, 256, H, H, OH, OH, 1.0f);
        conv3x3_direct2<<<(8 * 2 * OH * OH + 255) / 256, 256, 0, stream>>>(
            featAlt, 256, upflow, 2,
            (const float*)d_in[base + 1], (const float*)d_in[base + 2],
            flowAlt, 8, OH, OH, 2);
        { float* t = feat; feat = featAlt; featAlt = t; }
        { float* t = flow; flow = flowAlt; flowAlt = t; }
        H = OH;
    }

    // ---- final x4 upsample, scaled by 4.0 ----
    bilinear_ac_k<<<(8 * 2 * 256 * 256 + 255) / 256, 256, 0, stream>>>(
        flow, (float*)d_out, 8, 2, 64, 64, 256, 256, 4.0f);
}
